// PointNetSimple_1941325218382
// MI455X (gfx1250) — matrix operand order repaired and checked by reading, not compiled, not run on hardware
//
#include <hip/hip_runtime.h>

typedef _Float16 half_t;
typedef __attribute__((ext_vector_type(16))) _Float16 v16h;
typedef __attribute__((ext_vector_type(8)))  _Float16 v8h;
typedef __attribute__((ext_vector_type(8)))  float    v8f;

#define N_PTS 8192
#define KNN   16
#define NEDGE (N_PTS * KNN)   // 131072

// ---------------------------------------------------------------------------
// Kernel 1: exact kNN (K=16, self included). One wave32 per query point.
// Each lane scans 8192/32 = 256 candidates keeping a register-resident sorted
// top-16 (fully unrolled insertion => static indexing, no scratch). Lanes then
// merge 32x16 candidates through LDS with 16 rounds of wave-wide argmin.
// ---------------------------------------------------------------------------
__global__ void knn_kernel(const float* __restrict__ pos, int* __restrict__ idx_out) {
    const int lane = threadIdx.x & 31;
    const int wv   = threadIdx.x >> 5;            // 8 waves per block
    const int pt   = blockIdx.x * 8 + wv;         // one point per wave

    __shared__ float sd[8 * 512];
    __shared__ int   si[8 * 512];
    float* wsd = sd + wv * 512;
    int*   wsi = si + wv * 512;

    const float px = pos[pt * 3 + 0];
    const float py = pos[pt * 3 + 1];
    const float pz = pos[pt * 3 + 2];

    float bd[KNN];
    int   bi[KNN];
#pragma unroll
    for (int t = 0; t < KNN; ++t) { bd[t] = 3.0e38f; bi[t] = 0; }

    for (int j = lane; j < N_PTS; j += 32) {
        const float dx = pos[j * 3 + 0] - px;
        const float dy = pos[j * 3 + 1] - py;
        const float dz = pos[j * 3 + 2] - pz;
        float d = dx * dx + dy * dy + dz * dz;
        if (d < bd[KNN - 1]) {
            float cd = d; int ci = j;
#pragma unroll
            for (int t = 0; t < KNN; ++t) {
                if (cd < bd[t]) {
                    float td = bd[t]; int ti = bi[t];
                    bd[t] = cd; bi[t] = ci;
                    cd = td; ci = ti;
                }
            }
        }
    }

#pragma unroll
    for (int t = 0; t < KNN; ++t) {
        wsd[lane * KNN + t] = bd[t];
        wsi[lane * KNN + t] = bi[t];
    }
    __syncthreads();

    for (int r = 0; r < KNN; ++r) {
        float md = 3.0e38f; int mp = 0;
#pragma unroll
        for (int t = 0; t < 16; ++t) {
            const int p = lane + 32 * t;
            const float e = wsd[p];
            if (e < md) { md = e; mp = p; }
        }
#pragma unroll
        for (int off = 16; off >= 1; off >>= 1) {
            const float od = __shfl_xor(md, off, 32);
            const int   op = __shfl_xor(mp, off, 32);
            if (od < md || (od == md && op < mp)) { md = od; mp = op; }
        }
        if (lane == 0) {
            idx_out[pt * KNN + r] = wsi[mp];
            wsd[mp] = 3.0e38f;               // remove winner
        }
        __syncthreads();
    }
}

// ---------------------------------------------------------------------------
// Weight repack: f32 [KA x C] -> f16 WMMA B fragments, padded along K.
// Element g maps to (tile = ct*nChunks + kc, lane, t):
//   dst[((tile)*32 + lane)*16 + t] = W[kk*C + n], kk = kc*32 + (t>>3)*16 + (lane>>4)*8 + (t&7),
//   n = ct*16 + (lane&15); zero when kk >= KA. The GEMM then loads one aligned
//   32-byte v16h per lane per chunk.
// ---------------------------------------------------------------------------
__global__ void pack_kernel(const float* __restrict__ W, half_t* __restrict__ dst,
                            int KA, int C, int nChunks, int total) {
    const int g = blockIdx.x * 256 + threadIdx.x;
    if (g >= total) return;
    const int t    = g & 15;
    const int lane = (g >> 4) & 31;
    const int tile = g >> 9;
    const int kc   = tile % nChunks;
    const int ct   = tile / nChunks;
    const int kk   = kc * 32 + (t >> 3) * 16 + (lane >> 4) * 8 + (t & 7);
    const int n    = ct * 16 + (lane & 15);
    float w = 0.f;
    if (kk < KA) w = W[(size_t)kk * C + n];
    dst[g] = (half_t)w;
}

// ---------------------------------------------------------------------------
// Message staging: M[e, :] = [x_j , pos_j - pos_i , zero-pad] in f16,
// row-major with KPAD columns; rows built in registers, 16-byte v8h stores.
// ---------------------------------------------------------------------------
__global__ void gather1_kernel(const float* __restrict__ pos,
                               const int* __restrict__ idx,
                               half_t* __restrict__ M) {            // KPAD = 32
    const int e = blockIdx.x * 256 + threadIdx.x;
    if (e >= NEDGE) return;
    const int i = e >> 4;
    const int j = idx[e];
    const float jx = pos[j * 3 + 0], jy = pos[j * 3 + 1], jz = pos[j * 3 + 2];
    half_t* row = M + (size_t)e * 32;
    v8h r0 = {};
    r0[0] = (half_t)jx;
    r0[1] = (half_t)jy;
    r0[2] = (half_t)jz;
    r0[3] = (half_t)(jx - pos[i * 3 + 0]);
    r0[4] = (half_t)(jy - pos[i * 3 + 1]);
    r0[5] = (half_t)(jz - pos[i * 3 + 2]);
    const v8h z = {};
    *(v8h*)(row +  0) = r0;
    *(v8h*)(row +  8) = z;
    *(v8h*)(row + 16) = z;
    *(v8h*)(row + 24) = z;
}

__global__ void gather2_kernel(const float* __restrict__ pos,
                               const float* __restrict__ x,         // [N,64] f32
                               const int* __restrict__ idx,
                               half_t* __restrict__ M) {            // KPAD = 96
    const int e = blockIdx.x * 256 + threadIdx.x;
    if (e >= NEDGE) return;
    const int i = e >> 4;
    const int j = idx[e];
    half_t* row = M + (size_t)e * 96;
    const float4* xj4 = (const float4*)(x + (size_t)j * 64);   // 256B-aligned
#pragma unroll
    for (int c8 = 0; c8 < 8; ++c8) {
        const float4 a = xj4[c8 * 2 + 0];
        const float4 b = xj4[c8 * 2 + 1];
        v8h v;
        v[0] = (half_t)a.x; v[1] = (half_t)a.y; v[2] = (half_t)a.z; v[3] = (half_t)a.w;
        v[4] = (half_t)b.x; v[5] = (half_t)b.y; v[6] = (half_t)b.z; v[7] = (half_t)b.w;
        *(v8h*)(row + c8 * 8) = v;
    }
    v8h tail = {};
    tail[0] = (half_t)(pos[j * 3 + 0] - pos[i * 3 + 0]);
    tail[1] = (half_t)(pos[j * 3 + 1] - pos[i * 3 + 1]);
    tail[2] = (half_t)(pos[j * 3 + 2] - pos[i * 3 + 2]);
    const v8h z = {};
    *(v8h*)(row + 64) = tail;
    *(v8h*)(row + 72) = z;
    *(v8h*)(row + 80) = z;
    *(v8h*)(row + 88) = z;
}

// ---------------------------------------------------------------------------
// Fused PointNetConv: one workgroup per point, (CA/16) waves.
// Stage 1: H = relu(M @ Wa + ba)   [16 edges x CA]  -> LDS (f16)
// Stage 2: D = H @ Wb + bb; out[i,:] = relu(max over 16 edges)
// A fragments: two 16B loads/lane; B fragments: one 32B load/lane (pre-packed).
// ---------------------------------------------------------------------------
template <int CA, int COUT, int KPAD>
__global__ void pnconv_kernel(const half_t* __restrict__ M,
                              const half_t* __restrict__ pWa, const float* __restrict__ ba,
                              const half_t* __restrict__ pWb, const float* __restrict__ bb,
                              float* __restrict__ out) {
    __shared__ half_t Hs[16 * CA];

    const int lane    = threadIdx.x & 31;
    const int wv      = threadIdx.x >> 5;      // column-tile id for both stages
    const int i       = blockIdx.x;            // point id
    const int halfsel = lane >> 4;
    const int mrow    = lane & 15;
    const int klo     = halfsel * 8;
    const int n       = wv * 16 + mrow;        // output column

    // ---- stage 1: hidden activations ----
    {
        constexpr int NC = KPAD / 32;
        v8f acc = {};
#pragma unroll
        for (int kc = 0; kc < NC; ++kc) {
            const half_t* ap = M + ((size_t)i * 16 + mrow) * KPAD + kc * 32 + klo;
            const v8h a0 = *(const v8h*)ap;
            const v8h a1 = *(const v8h*)(ap + 16);
            v16h A;
#pragma unroll
            for (int t = 0; t < 8; ++t) { A[t] = a0[t]; A[8 + t] = a1[t]; }
            const v16h B = *(const v16h*)(pWa + ((size_t)(wv * NC + kc) * 32 + lane) * 16);
            acc = __builtin_amdgcn_wmma_f32_16x16x32_f16(false, A, false, B,
                                                         (short)0, acc, false, false);
        }
        const float bias = ba[n];
#pragma unroll
        for (int v = 0; v < 8; ++v) {
            const int row = v + halfsel * 8;
            const float h = acc[v] + bias;
            Hs[row * CA + n] = (half_t)fmaxf(h, 0.f);
        }
    }
    __syncthreads();

    // ---- stage 2: output GEMM + max over the 16 edges + bias + relu ----
    {
        constexpr int NC = CA / 32;
        v8f acc = {};
#pragma unroll
        for (int kc = 0; kc < NC; ++kc) {
            const half_t* ap = Hs + mrow * CA + kc * 32 + klo;
            const v8h a0 = *(const v8h*)ap;
            const v8h a1 = *(const v8h*)(ap + 16);
            v16h A;
#pragma unroll
            for (int t = 0; t < 8; ++t) { A[t] = a0[t]; A[8 + t] = a1[t]; }
            const v16h B = *(const v16h*)(pWb + ((size_t)(wv * NC + kc) * 32 + lane) * 16);
            acc = __builtin_amdgcn_wmma_f32_16x16x32_f16(false, A, false, B,
                                                         (short)0, acc, false, false);
        }
        float m = acc[0];
#pragma unroll
        for (int v = 1; v < 8; ++v) m = fmaxf(m, acc[v]);
        m = fmaxf(m, __shfl_xor(m, 16, 32));   // other half of the 16 edge rows
        m = fmaxf(m + bb[n], 0.f);
        if (lane < 16) out[(size_t)i * COUT + n] = m;
    }
}

// ---------------------------------------------------------------------------
// Launch: weight packs + knn -> [gather, fused conv] x 3.
// Workspace: idx (512 KB) + f16 messages (24 MB) + packed weights (~90 KB).
// ---------------------------------------------------------------------------
extern "C" void kernel_launch(void* const* d_in, const int* in_sizes, int n_in,
                              void* d_out, int out_size, void* d_ws, size_t ws_size,
                              hipStream_t stream) {
    (void)in_sizes; (void)n_in; (void)out_size; (void)ws_size;

    const float* pos = (const float*)d_in[0];
    const float* W1a = (const float*)d_in[1];  const float* b1a = (const float*)d_in[2];
    const float* W1b = (const float*)d_in[3];  const float* b1b = (const float*)d_in[4];
    const float* W2a = (const float*)d_in[5];  const float* b2a = (const float*)d_in[6];
    const float* W2b = (const float*)d_in[7];  const float* b2b = (const float*)d_in[8];
    const float* W3a = (const float*)d_in[9];  const float* b3a = (const float*)d_in[10];
    const float* W3b = (const float*)d_in[11]; const float* b3b = (const float*)d_in[12];

    float* out = (float*)d_out;
    float* h1 = out;                       // [8192, 64]
    float* h2 = out + (size_t)N_PTS * 64;  // [8192, 64]
    float* h3 = h2  + (size_t)N_PTS * 64;  // [8192, 128]

    char* ws = (char*)d_ws;
    size_t off = 0;
    int* idx = (int*)ws;                       off += ((size_t)NEDGE * 4 + 255) & ~(size_t)255;
    half_t* M = (half_t*)(ws + off);           off += (size_t)NEDGE * 96 * 2;
    // packed weight fragments (f16): sizes = CT * nChunks * 512 halves
    half_t* pW1a = (half_t*)(ws + off);  const int n1a = 4 * 1 * 512;  off += (size_t)n1a * 2;
    half_t* pW1b = (half_t*)(ws + off);  const int n1b = 4 * 2 * 512;  off += (size_t)n1b * 2;
    half_t* pW2a = (half_t*)(ws + off);  const int n2a = 4 * 3 * 512;  off += (size_t)n2a * 2;
    half_t* pW2b = (half_t*)(ws + off);  const int n2b = 4 * 2 * 512;  off += (size_t)n2b * 2;
    half_t* pW3a = (half_t*)(ws + off);  const int n3a = 8 * 3 * 512;  off += (size_t)n3a * 2;
    half_t* pW3b = (half_t*)(ws + off);  const int n3b = 8 * 4 * 512;  off += (size_t)n3b * 2;

    // 0) pre-pack all weight matrices into WMMA B-fragment layout
    pack_kernel<<<(n1a + 255) / 256, 256, 0, stream>>>(W1a, pW1a,   6,  64, 1, n1a);
    pack_kernel<<<(n1b + 255) / 256, 256, 0, stream>>>(W1b, pW1b,  64,  64, 2, n1b);
    pack_kernel<<<(n2a + 255) / 256, 256, 0, stream>>>(W2a, pW2a,  67,  64, 3, n2a);
    pack_kernel<<<(n2b + 255) / 256, 256, 0, stream>>>(W2b, pW2b,  64,  64, 2, n2b);
    pack_kernel<<<(n3a + 255) / 256, 256, 0, stream>>>(W3a, pW3a,  67, 128, 3, n3a);
    pack_kernel<<<(n3b + 255) / 256, 256, 0, stream>>>(W3b, pW3b, 128, 128, 4, n3b);

    // 1) exact kNN
    knn_kernel<<<N_PTS / 8, 256, 0, stream>>>(pos, idx);

    // 2) layer 1: 6 -> 64 -> 64
    gather1_kernel<<<NEDGE / 256, 256, 0, stream>>>(pos, idx, M);
    pnconv_kernel<64, 64, 32><<<N_PTS, 128, 0, stream>>>(M, pW1a, b1a, pW1b, b1b, h1);

    // 3) layer 2: 67 -> 64 -> 64
    gather2_kernel<<<NEDGE / 256, 256, 0, stream>>>(pos, h1, idx, M);
    pnconv_kernel<64, 64, 96><<<N_PTS, 128, 0, stream>>>(M, pW2a, b2a, pW2b, b2b, h2);

    // 4) layer 3: 67 -> 128 -> 128
    gather2_kernel<<<NEDGE / 256, 256, 0, stream>>>(pos, h2, idx, M);
    pnconv_kernel<128, 128, 96><<<N_PTS, 256, 0, stream>>>(M, pW3a, b3a, pW3b, b3b, h3);
}
